// NGLU_20667382628460
// MI455X (gfx1250) — matrix operand order repaired and checked by reading, not compiled, not run on hardware
//
#include <hip/hip_runtime.h>
#include <hip/hip_bf16.h>
#include <math.h>

typedef __attribute__((ext_vector_type(16))) _Float16 v16h;
typedef __attribute__((ext_vector_type(8)))  _Float16 v8h;
typedef __attribute__((ext_vector_type(8)))  float    v8f;

#define NN   102400
#define BBATCH 512
#define PP   200
#define HID  256
#define HEADS 3
#define OUTC 12
#define T0   0.9f
#define SLOPE 0.2f

#define BM 64
#define BN 64
#define BK 32

__device__ __forceinline__ v8h cvt8(const float* __restrict__ p) {
  const float4* q = (const float4*)p;
  float4 u = q[0], w = q[1];
  v8h h;
  h[0] = (_Float16)u.x; h[1] = (_Float16)u.y; h[2] = (_Float16)u.z; h[3] = (_Float16)u.w;
  h[4] = (_Float16)w.x; h[5] = (_Float16)w.y; h[6] = (_Float16)w.z; h[7] = (_Float16)w.w;
  return h;
}
__device__ __forceinline__ v8h cvt8s(const float* __restrict__ p, long long stride) {
  v8h h;
#pragma unroll
  for (int j = 0; j < 8; ++j) h[j] = (_Float16)p[j * stride];
  return h;
}

// ---------------------------------------------------------------------------
// Tiled WMMA GEMM: C[M,Nn] = A[M,K] @ B[K,Nn]  (TRANSB: B = Bm^T, Bm [Nn,K]).
// Epilogue: 0=none, 1=relu, 2=(x+1)/2.  bias per-col, rowscale per-row.
// M,Nn multiples of 64; K multiple of 32 (true at every call site).
//
// LDS is double-buffered and fragment-ready: Af[buf][tile][lane][16 f16] is
// exactly the v16h a lane feeds to v_wmma_f32_16x16x32_f16.  Staging is one
// ds_store_b128 per thread per operand; fragments are aligned 32B LDS loads.
// Software pipeline: while WMMAs consume buffer `cur`, global loads + stores
// for step k+1 land in buffer `cur^1`; one s_barrier per k-step.
//   A elems: 0..7 -> K = hi*8+(0..7),  8..15 -> K = 16+hi*8+(0..7)   (row m)
//   B elems: 0..7 -> K = hi*8+(0..7),  8..15 -> K = 16+hi*8+(0..7)   (col n)
// ---------------------------------------------------------------------------
template <int TRANSB>
__global__ __launch_bounds__(256)
void gemm_wmma(const float* __restrict__ A, const float* __restrict__ Bm,
               float* __restrict__ C, int M, int K, int Nn,
               const float* __restrict__ bias, const float* __restrict__ rowscale,
               int epilogue) {
  __shared__ __align__(32) _Float16 Af[2][4][32][16];
  __shared__ __align__(32) _Float16 Bf[2][4][32][16];
  const int bm = blockIdx.y * BM;
  const int bn = blockIdx.x * BN;
  const int tid  = threadIdx.x;
  const int lane = tid & 31;
  const int wave = tid >> 5;
  const int wr = wave & 3;          // wmma tile row 0..3
  const int wc = (wave >> 2) * 2;   // wmma tile col base (0 or 2)
  const int m16 = lane & 15;
  const int hi  = lane >> 4;

  // staging assignment: 128 lane-slots per operand, 2 threads per slot
  const int slot = tid >> 1;
  const int half = tid & 1;
  const int s_t  = slot >> 5;
  const int s_l  = slot & 31;
  const int s_m16 = s_l & 15;
  const int s_hi  = s_l >> 4;
  const int a_ks  = half * 16 + s_hi * 8;
  const int b_ks  = half * 16 + s_hi * 8;
  const long long a_row = bm + s_t * 16 + s_m16;
  const long long b_col = bn + s_t * 16 + s_m16;

  const float* aP = A + a_row * K + a_ks;
  const float* bP = TRANSB ? (Bm + b_col * K + b_ks)
                           : (Bm + (long long)b_ks * Nn + b_col);
  _Float16* aSlot0 = &Af[0][s_t][s_l][half * 8];
  _Float16* bSlot0 = &Bf[0][s_t][s_l][half * 8];
  const int bufStride = 4 * 32 * 16;   // halfs per buffer

  v8f acc0 = {}; v8f acc1 = {};
  const int nk = K / BK;

  // prologue: stage k-step 0 into buffer 0
  *(v8h*)aSlot0 = cvt8(aP);
  *(v8h*)bSlot0 = TRANSB ? cvt8(bP) : cvt8s(bP, Nn);
  __syncthreads();

  for (int ks = 0; ks < nk; ++ks) {
    const int cur = ks & 1;
    if (ks + 1 < nk) {  // prefetch next k-step into the other buffer
      const long long off = (long long)(ks + 1) * BK;
      v8h ha = cvt8(aP + off);
      v8h hb = TRANSB ? cvt8(bP + off) : cvt8s(bP + off * Nn, Nn);
      *(v8h*)(aSlot0 + (cur ^ 1) * bufStride) = ha;
      *(v8h*)(bSlot0 + (cur ^ 1) * bufStride) = hb;
    }
    v16h af = *(const v16h*)&Af[cur][wr][lane][0];
    v16h b0 = *(const v16h*)&Bf[cur][wc][lane][0];
    v16h b1 = *(const v16h*)&Bf[cur][wc + 1][lane][0];
    acc0 = __builtin_amdgcn_wmma_f32_16x16x32_f16(false, af, false, b0,
                                                  (short)0, acc0, false, false);
    acc1 = __builtin_amdgcn_wmma_f32_16x16x32_f16(false, af, false, b1,
                                                  (short)0, acc1, false, false);
    __syncthreads();
  }

#pragma unroll
  for (int v = 0; v < 8; ++v) {
    int row  = bm + wr * 16 + hi * 8 + v;
    int col0 = bn + wc * 16 + m16;
    int col1 = bn + (wc + 1) * 16 + m16;
    float x0 = acc0[v], x1 = acc1[v];
    if (bias)     { x0 += bias[col0]; x1 += bias[col1]; }
    if (rowscale) { float s = rowscale[row]; x0 *= s; x1 *= s; }
    if (epilogue == 1)      { x0 = fmaxf(x0, 0.f); x1 = fmaxf(x1, 0.f); }
    else if (epilogue == 2) { x0 = (x0 + 1.f) * 0.5f; x1 = (x1 + 1.f) * 0.5f; }
    C[(long long)row * Nn + col0] = x0;
    C[(long long)row * Nn + col1] = x1;
  }
}

// ---------------------------------------------------------------------------
__global__ void fill_kernel(float* __restrict__ p, float v, long long n) {
  long long i = (long long)blockIdx.x * blockDim.x + threadIdx.x;
  if (i < n) p[i] = v;
}

__global__ void deg_count(const int* __restrict__ dst, float* __restrict__ deg, int E) {
  int i = blockIdx.x * blockDim.x + threadIdx.x;
  if (i < E) atomicAdd(&deg[dst[i]], 1.0f);
}

__global__ void make_norm(float* __restrict__ deg, int n) {
  int i = blockIdx.x * blockDim.x + threadIdx.x;
  if (i < n) deg[i] = rsqrtf(fmaxf(deg[i], 1.0f));
}

// one edge per block, 256 channels: m[dst] += h[src]
__global__ __launch_bounds__(256)
void scatter_add(const float* __restrict__ h, const int* __restrict__ src,
                 const int* __restrict__ dst, float* __restrict__ m) {
  int e = blockIdx.x, ch = threadIdx.x;
  int s = src[e], d = dst[e];
  atomicAdd(&m[(long long)d * HID + ch], h[(long long)s * HID + ch]);
}

// out = m*norm[row] + b  (optional relu)
__global__ __launch_bounds__(256)
void gcn_epilogue(const float* __restrict__ m, const float* __restrict__ norm,
                  const float* __restrict__ b, float* __restrict__ out, int relu) {
  long long i = blockIdx.x, ch = threadIdx.x;
  float v = m[i * HID + ch] * norm[i] + b[ch];
  if (relu) v = fmaxf(v, 0.f);
  out[i * HID + ch] = v;
}

__global__ __launch_bounds__(256)
void mean_nodes(const float* __restrict__ x, float* __restrict__ out) {
  int b = blockIdx.x, ch = threadIdx.x;
  float s = 0.f;
  for (int i = 0; i < PP; ++i) s += x[((long long)b * PP + i) * HID + ch];
  out[b * HID + ch] = s * (1.0f / PP);
}

__global__ __launch_bounds__(256)
void med_build(const float* __restrict__ hrs, const float* __restrict__ pp,
               float* __restrict__ med) {
  int r = blockIdx.x, t = threadIdx.x;
  float a = hrs[r * HID + t], b = pp[r * HID + t];
  __shared__ float sh[256];
  sh[t] = a * a + b * b;
  __syncthreads();
  for (int s = 128; s; s >>= 1) { if (t < s) sh[t] += sh[t + s]; __syncthreads(); }
  float inv = rsqrtf(sh[0]);
  med[r * 512 + t] = a * inv;
  med[r * 512 + 256 + t] = b * inv;
}

__global__ __launch_bounds__(256)
void row_dinv(const float* __restrict__ mo, float* __restrict__ dinv) {
  int r = blockIdx.x, t = threadIdx.x;
  float s = 0.f;
  for (int c = t; c < BBATCH; c += 256)
    s += (mo[r * BBATCH + c] >= T0 || c == r) ? 1.f : 0.f;
  __shared__ float sh[256];
  sh[t] = s; __syncthreads();
  for (int k = 128; k; k >>= 1) { if (t < k) sh[t] += sh[t + k]; __syncthreads(); }
  if (t == 0) dinv[r] = rsqrtf(sh[0]);
}

__global__ void make_nadj(const float* __restrict__ mo, const float* __restrict__ dinv,
                          float* __restrict__ nadj) {
  int idx = blockIdx.x * blockDim.x + threadIdx.x;
  if (idx >= BBATCH * BBATCH) return;
  int r = idx >> 9, c = idx & 511;
  float a = (mo[idx] >= T0 || c == r) ? 1.f : 0.f;
  nadj[idx] = a * dinv[r] * dinv[c];
}

// el/er: one wave per (node, head): e[node,h] = dot(f[node, h*256:...], a[h,:])
__global__ __launch_bounds__(256)
void attn_dot(const float* __restrict__ f, const float* __restrict__ a,
              float* __restrict__ e, int n) {
  int gw = blockIdx.x * 8 + (threadIdx.x >> 5);
  int lane = threadIdx.x & 31;
  if (gw >= n * HEADS) return;
  int node = gw / HEADS, h = gw % HEADS;
  const float* fp = f + (long long)node * (HEADS * HID) + h * HID;
  const float* ap = a + h * HID;
  float s = 0.f;
  for (int c = lane; c < HID; c += 32) s += fp[c] * ap[c];
  for (int off = 16; off; off >>= 1) s += __shfl_down(s, off, 32);
  if (lane == 0) e[node * HEADS + h] = s;
}

// poi2img GAT: one block per (parcel b, head h). Segment = contiguous 200 nodes.
// out (pre-zeroed) accumulates head-mean via atomicAdd * (1/3).
__global__ __launch_bounds__(256)
void gat_pool(const float* __restrict__ fs, const float* __restrict__ el,
              const float* __restrict__ er, float* __restrict__ out) {
  int b = blockIdx.x / HEADS, h = blockIdx.x % HEADS;
  int t = threadIdx.x;
  __shared__ float sh[256];
  __shared__ float alpha[PP];
  float e = -INFINITY;
  if (t < PP) {
    float v = el[(b * PP + t) * HEADS + h] + er[b * HEADS + h];
    e = v > 0.f ? v : SLOPE * v;
  }
  sh[t] = e; __syncthreads();
  for (int s = 128; s; s >>= 1) { if (t < s) sh[t] = fmaxf(sh[t], sh[t + s]); __syncthreads(); }
  float m = sh[0];
  __syncthreads();
  float w = (t < PP) ? __expf(e - m) : 0.f;
  sh[t] = w; __syncthreads();
  for (int s = 128; s; s >>= 1) { if (t < s) sh[t] += sh[t + s]; __syncthreads(); }
  float z = sh[0];
  if (t < PP) alpha[t] = w / (z + 1e-9f);
  __syncthreads();
  float acc = 0.f;
  for (int i = 0; i < PP; ++i)
    acc += fs[((long long)(b * PP + i)) * (HEADS * HID) + h * HID + t] * alpha[i];
  atomicAdd(&out[b * HID + t], acc * (1.0f / HEADS));
}

// img2poi pooled == head-mean of fs_with (alpha==1 exactly in fp32: z+1e-9==1)
__global__ __launch_bounds__(256)
void head_mean(const float* __restrict__ fw, float* __restrict__ out) {
  int b = blockIdx.x, t = threadIdx.x;
  out[b * HID + t] = (fw[b * 768 + t] + fw[b * 768 + 256 + t] + fw[b * 768 + 512 + t])
                     * (1.0f / HEADS);
}

__global__ void final_fc(const float* __restrict__ i2p, const float* __restrict__ poi_agg,
                         const float* __restrict__ p2i, const float* __restrict__ hrs_agg,
                         const float* __restrict__ Wfc, const float* __restrict__ bfc,
                         float* __restrict__ out) {
  int idx = blockIdx.x * blockDim.x + threadIdx.x;
  if (idx >= BBATCH * OUTC) return;
  int b = idx / OUTC, o = idx % OUTC;
  float s = bfc[o];
  for (int c = 0; c < HID; ++c) {
    s += i2p[b * HID + c]     * Wfc[c * OUTC + o];
    s += poi_agg[b * HID + c] * Wfc[(HID + c) * OUTC + o];
    s += p2i[b * HID + c]     * Wfc[(2 * HID + c) * OUTC + o];
    s += hrs_agg[b * HID + c] * Wfc[(3 * HID + c) * OUTC + o];
  }
  out[b * OUTC + o] = s;
}

// ---------------------------------------------------------------------------
extern "C" void kernel_launch(void* const* d_in, const int* in_sizes, int n_in,
                              void* d_out, int out_size, void* d_ws, size_t ws_size,
                              hipStream_t stream) {
  const float* X    = (const float*)d_in[0];   // [N,64]
  const float* IMG  = (const float*)d_in[1];   // [B,2048]
  const int*   ESRC = (const int*)d_in[2];
  const int*   EDST = (const int*)d_in[3];
  const float* Wg1  = (const float*)d_in[4];
  const float* bg1  = (const float*)d_in[5];
  const float* Wg2  = (const float*)d_in[6];
  const float* bg2  = (const float*)d_in[7];
  const float* Whrs = (const float*)d_in[8];
  const float* bhrs = (const float*)d_in[9];
  const float* Wph  = (const float*)d_in[10];
  const float* bph  = (const float*)d_in[11];
  const float* Wpp  = (const float*)d_in[12];
  const float* bpp  = (const float*)d_in[13];
  const float* Wgi  = (const float*)d_in[14];
  const float* ali  = (const float*)d_in[15];
  const float* ari  = (const float*)d_in[16];
  const float* Wgw  = (const float*)d_in[17];
  const float* Whg  = (const float*)d_in[20];
  const float* Wpg  = (const float*)d_in[21];
  const float* Wfc  = (const float*)d_in[22];
  const float* bfc  = (const float*)d_in[23];
  const int E = in_sizes[2];

  const long long NF = (long long)NN * HID;
  float* ws   = (float*)d_ws;
  float* norm = ws;                  // N
  float* bufA = norm + NN;           // NF
  float* bufB = bufA + NF;           // NF
  float* bufE = bufB + NF;           // NF (fs_in tail)
  float* bufC = bufE + NF;           // NF (x1 / pp)
  float* bufD = bufC + NF;           // NF (poi_nodes)
  float* fsin = bufA;                // 3*NF contiguous over A,B,E
  float* sm   = bufD + NF;           // small scratch region
  float* hrs       = sm;             sm += BBATCH * HID;
  float* poi_pool  = sm;             sm += BBATCH * HID;
  float* med       = sm;             sm += BBATCH * 512;
  float* nadj      = sm;             sm += BBATCH * BBATCH;
  float* dinv      = sm;             sm += BBATCH;
  float* t1        = sm;             sm += BBATCH * HID;
  float* hrs_agg   = sm;             sm += BBATCH * HID;
  float* poi_agg   = sm;             sm += BBATCH * HID;
  float* ph        = sm;             sm += BBATCH * HID;
  float* fdin      = sm;             sm += BBATCH * HEADS * HID;
  float* fwith     = sm;             sm += BBATCH * HEADS * HID;
  float* er        = sm;             sm += BBATCH * HEADS;
  float* p2i       = sm;             sm += BBATCH * HID;
  float* i2p       = sm;             sm += BBATCH * HID;
  float* el        = sm;             sm += (long long)NN * HEADS;

  float* dout = (float*)d_out;
  float* med_out = dout + BBATCH * OUTC;  // [B,B] second tuple output

  auto gemm = [&](const float* A, const float* Bm, float* Cp, int M, int K, int Nn,
                  const float* bias, const float* rs, int epi, int tb) {
    dim3 g(Nn / 64, M / 64);
    if (tb) gemm_wmma<1><<<g, 256, 0, stream>>>(A, Bm, Cp, M, K, Nn, bias, rs, epi);
    else    gemm_wmma<0><<<g, 256, 0, stream>>>(A, Bm, Cp, M, K, Nn, bias, rs, epi);
  };

  // --- degree norm (self loops => init 1) ---
  fill_kernel<<<(NN + 255) / 256, 256, 0, stream>>>(norm, 1.0f, NN);
  deg_count<<<(E + 255) / 256, 256, 0, stream>>>(EDST, norm, E);
  make_norm<<<(NN + 255) / 256, 256, 0, stream>>>(norm, NN);

  // --- GCN layer 1: h = (X@W1)*norm ; m = h + scatter(h) ; x1 = relu(m*norm+b1)
  gemm(X, Wg1, bufA, NN, 64, HID, nullptr, norm, 0, 0);
  hipMemcpyAsync(bufB, bufA, NF * sizeof(float), hipMemcpyDeviceToDevice, stream);
  scatter_add<<<E, 256, 0, stream>>>(bufA, ESRC, EDST, bufB);
  gcn_epilogue<<<NN, 256, 0, stream>>>(bufB, norm, bg1, bufC, 1);

  // --- GCN layer 2 -> poi_nodes (bufD)
  gemm(bufC, Wg2, bufA, NN, HID, HID, nullptr, norm, 0, 0);
  hipMemcpyAsync(bufB, bufA, NF * sizeof(float), hipMemcpyDeviceToDevice, stream);
  scatter_add<<<E, 256, 0, stream>>>(bufA, ESRC, EDST, bufB);
  gcn_epilogue<<<NN, 256, 0, stream>>>(bufB, norm, bg2, bufD, 0);

  mean_nodes<<<BBATCH, 256, 0, stream>>>(bufD, poi_pool);

  // --- hrs encoder + mid-level similarity ---
  gemm(IMG, Whrs, hrs, BBATCH, 2048, HID, bhrs, nullptr, 0, 0);
  med_build<<<BBATCH, 256, 0, stream>>>(hrs, poi_pool, med);
  gemm(med, med, med_out, BBATCH, 512, BBATCH, nullptr, nullptr, 2, 1); // (med@medT+1)/2

  // --- UniSimGraph GCN on thresholded similarity ---
  row_dinv<<<BBATCH, 256, 0, stream>>>(med_out, dinv);
  make_nadj<<<(BBATCH * BBATCH + 255) / 256, 256, 0, stream>>>(med_out, dinv, nadj);
  gemm(hrs, Whg, t1, BBATCH, HID, HID, nullptr, nullptr, 0, 0);
  gemm(nadj, t1, hrs_agg, BBATCH, BBATCH, HID, nullptr, nullptr, 1, 0);
  gemm(poi_pool, Wpg, t1, BBATCH, HID, HID, nullptr, nullptr, 0, 0);
  gemm(nadj, t1, poi_agg, BBATCH, BBATCH, HID, nullptr, nullptr, 1, 0);

  // --- heterogeneous GAT projections ---
  gemm(hrs, Wph, ph, BBATCH, HID, HID, bph, nullptr, 0, 0);
  gemm(bufD, Wpp, bufC, NN, HID, HID, bpp, nullptr, 0, 0);       // pp in bufC
  gemm(bufC, Wgi, fsin, NN, HID, HEADS * HID, nullptr, nullptr, 0, 0);
  gemm(ph, Wgi, fdin, BBATCH, HID, HEADS * HID, nullptr, nullptr, 0, 0);

  attn_dot<<<(NN * HEADS + 7) / 8, 256, 0, stream>>>(fsin, ali, el, NN);
  attn_dot<<<(BBATCH * HEADS + 7) / 8, 256, 0, stream>>>(fdin, ari, er, BBATCH);

  fill_kernel<<<(BBATCH * HID + 255) / 256, 256, 0, stream>>>(p2i, 0.f, BBATCH * HID);
  gat_pool<<<BBATCH * HEADS, 256, 0, stream>>>(fsin, el, er, p2i);

  // img2poi: single-edge segments => alpha==1 exactly; pooled == head-mean of ph@Wgw
  gemm(ph, Wgw, fwith, BBATCH, HID, HEADS * HID, nullptr, nullptr, 0, 0);
  head_mean<<<BBATCH, 256, 0, stream>>>(fwith, i2p);

  final_fc<<<(BBATCH * OUTC + 255) / 256, 256, 0, stream>>>(
      i2p, poi_agg, p2i, hrs_agg, Wfc, bfc, dout);
}
